// GptOssDecoderLayer_5196910428359
// MI455X (gfx1250) — compile-verified, matrix operand order checked by reading, not run on hardware
//
#include <hip/hip_runtime.h>
#include <hip/hip_bf16.h>

typedef __attribute__((ext_vector_type(16))) _Float16 v16h;
typedef __attribute__((ext_vector_type(8)))  float    v8f;
typedef __attribute__((ext_vector_type(4)))  _Float16 h4;

union F16Frag { v16h v; uint4 u[2]; _Float16 h[16]; };
union F16x8   { uint4 u; _Float16 h[8]; };

#define S_LEN   2048
#define DMODEL  2048
#define NHEADS  32
#define NKV     8
#define HDIM    64
#define FFN_DIM 8192
#define EPS_V   1e-5f
#define NEG_V   -1000000000.0f

static __device__ __forceinline__ v8f vzero() {
    v8f z;
#pragma unroll
    for (int i = 0; i < 8; ++i) z[i] = 0.0f;
    return z;
}

static __device__ __forceinline__ v8f wmma16(const F16Frag& a, const F16Frag& b, v8f c) {
    // D = A(16x32 f16) x B(32x16 f16) + C(f32)
    return __builtin_amdgcn_wmma_f32_16x16x32_f16(false, a.v, false, b.v, (short)0, c, false, false);
}

// Async global->LDS 16B copy (CDNA5, ASYNCcnt-tracked). lds: generic pointer to
// __shared__ (low 32 bits are the LDS byte offset); g: per-lane global address.
static __device__ __forceinline__ void async_b128(void* lds, const void* g) {
    unsigned l = (unsigned)(size_t)lds;
    asm volatile("global_load_async_to_lds_b128 %0, %1, off"
                 :: "v"(l), "v"(g)
                 : "memory");
}
static __device__ __forceinline__ void wait_async0() {
    asm volatile("s_wait_asynccnt 0" ::: "memory");
}

// ---------------------------------------------------------------------------
// RMSNorm: one block (256 threads) per row of 2048 fp32; writes f16.
// ---------------------------------------------------------------------------
__global__ __launch_bounds__(256) void rmsnorm_kernel(const float* __restrict__ x,
                                                      const float* __restrict__ w,
                                                      _Float16* __restrict__ out) {
    const int row = blockIdx.x;
    const int t = threadIdx.x;
    const float* xr = x + (size_t)row * DMODEL + t * 8;
    float4 a = *(const float4*)(xr);
    float4 b = *(const float4*)(xr + 4);
    float s = a.x*a.x + a.y*a.y + a.z*a.z + a.w*a.w
            + b.x*b.x + b.y*b.y + b.z*b.z + b.w*b.w;
#pragma unroll
    for (int off = 16; off >= 1; off >>= 1) s += __shfl_xor(s, off);
    __shared__ float red[8];
    if ((t & 31) == 0) red[t >> 5] = s;
    __syncthreads();
    float tot = 0.0f;
#pragma unroll
    for (int i = 0; i < 8; ++i) tot += red[i];
    const float rs = rsqrtf(tot * (1.0f / DMODEL) + EPS_V);
    const float* wr = w + t * 8;
    float4 wa = *(const float4*)(wr);
    float4 wb = *(const float4*)(wr + 4);
    F16x8 o;
    o.h[0] = (_Float16)(a.x * rs * wa.x); o.h[1] = (_Float16)(a.y * rs * wa.y);
    o.h[2] = (_Float16)(a.z * rs * wa.z); o.h[3] = (_Float16)(a.w * rs * wa.w);
    o.h[4] = (_Float16)(b.x * rs * wb.x); o.h[5] = (_Float16)(b.y * rs * wb.y);
    o.h[6] = (_Float16)(b.z * rs * wb.z); o.h[7] = (_Float16)(b.w * rs * wb.w);
    *(uint4*)(out + (size_t)row * DMODEL + t * 8) = o.u;
}

// ---------------------------------------------------------------------------
// Generic WMMA GEMM: C[M,N] = A[M,K](f16) @ Bw[N,K](f32->f16)^T + bias
// Tile 128x128, 8 waves (2x4), wave tile 64x32, K-step 32.
// Double-buffered LDS; A tiles staged with async global->LDS copies, B tiles
// staged through VGPRs (fp32->f16 convert) with loads issued before the WMMAs.
// MODE 0: f16 out row-major.  MODE 1: f16 out transposed (C[N,M]).
// MODE 2: f32 out = resid + acc + bias.
// ---------------------------------------------------------------------------
template <int MODE>
__global__ __launch_bounds__(256) void gemm_kernel(const _Float16* __restrict__ A,
                                                   const float* __restrict__ Bw,
                                                   const float* __restrict__ bias,
                                                   void* __restrict__ Cout,
                                                   const float* __restrict__ resid,
                                                   int N, int K) {
    __shared__ _Float16 Asm[2][128][40];
    __shared__ _Float16 Bsm[2][128][40];
    const int t = threadIdx.x;
    const int lane = t & 31;
    const int wav = t >> 5;
    const int wm = wav >> 2;       // 0..1
    const int wn = wav & 3;        // 0..3
    const int m0 = blockIdx.y * 128;
    const int n0 = blockIdx.x * 128;
    const int r  = lane & 15;
    const int hi = (lane < 16) ? 0 : 8;

    // Per-thread staging coordinates.
    const int arow = t >> 2, akc = (t & 3) * 8;   // + i*64 rows, 2 chunks
    const int brow = t >> 3, bkc = (t & 7) * 4;   // + i*32 rows, 4 chunks

    v8f acc[4][2];
#pragma unroll
    for (int mt = 0; mt < 4; ++mt)
#pragma unroll
        for (int nt = 0; nt < 2; ++nt) acc[mt][nt] = vzero();

    const int nk = K >> 5;

    // Prologue: stage tile 0 into buffer 0.
#pragma unroll
    for (int i = 0; i < 2; ++i)
        async_b128(&Asm[0][arow + i*64][akc],
                   A + (size_t)(m0 + arow + i*64) * K + akc);
#pragma unroll
    for (int i = 0; i < 4; ++i) {
        float4 f = *(const float4*)(Bw + (size_t)(n0 + brow + i*32) * K + bkc);
        h4 hv;
        hv[0] = (_Float16)f.x; hv[1] = (_Float16)f.y;
        hv[2] = (_Float16)f.z; hv[3] = (_Float16)f.w;
        *(h4*)&Bsm[0][brow + i*32][bkc] = hv;
    }
    wait_async0();
    __syncthreads();

    for (int kk = 0; kk < nk; ++kk) {
        const int cur = kk & 1;
        const int nxt = cur ^ 1;
        const bool more = (kk + 1) < nk;
        const int k1 = (kk + 1) << 5;

        F16Frag af[4], bf[2];
#pragma unroll
        for (int mt = 0; mt < 4; ++mt) {
            int row = wm * 64 + mt * 16 + r;
            af[mt].u[0] = *(const uint4*)&Asm[cur][row][hi];
            af[mt].u[1] = *(const uint4*)&Asm[cur][row][hi + 16];
        }
#pragma unroll
        for (int nt = 0; nt < 2; ++nt) {
            int row = wn * 32 + nt * 16 + r;
            bf[nt].u[0] = *(const uint4*)&Bsm[cur][row][hi];
            bf[nt].u[1] = *(const uint4*)&Bsm[cur][row][hi + 16];
        }

        // Kick off next tile's traffic before the MMAs.
        float4 breg[4];
        if (more) {
#pragma unroll
            for (int i = 0; i < 2; ++i)
                async_b128(&Asm[nxt][arow + i*64][akc],
                           A + (size_t)(m0 + arow + i*64) * K + k1 + akc);
#pragma unroll
            for (int i = 0; i < 4; ++i)
                breg[i] = *(const float4*)(Bw + (size_t)(n0 + brow + i*32) * K + k1 + bkc);
        }

#pragma unroll
        for (int mt = 0; mt < 4; ++mt)
#pragma unroll
            for (int nt = 0; nt < 2; ++nt)
                acc[mt][nt] = wmma16(af[mt], bf[nt], acc[mt][nt]);

        if (more) {
#pragma unroll
            for (int i = 0; i < 4; ++i) {
                h4 hv;
                hv[0] = (_Float16)breg[i].x; hv[1] = (_Float16)breg[i].y;
                hv[2] = (_Float16)breg[i].z; hv[3] = (_Float16)breg[i].w;
                *(h4*)&Bsm[nxt][brow + i*32][bkc] = hv;
            }
        }
        wait_async0();
        __syncthreads();
    }

    // Epilogue
#pragma unroll
    for (int mt = 0; mt < 4; ++mt) {
#pragma unroll
        for (int nt = 0; nt < 2; ++nt) {
            const int n = n0 + wn * 32 + nt * 16 + r;
            const float bn = bias[n];
            const int mbase = m0 + wm * 64 + mt * 16 + hi;
            if (MODE == 0) {
                _Float16* C = (_Float16*)Cout;
#pragma unroll
                for (int i = 0; i < 8; ++i)
                    C[(size_t)(mbase + i) * N + n] = (_Float16)(acc[mt][nt][i] + bn);
            } else if (MODE == 1) {
                _Float16* C = (_Float16*)Cout;
                const int M = gridDim.y * 128;
                F16x8 fr;
#pragma unroll
                for (int i = 0; i < 8; ++i) fr.h[i] = (_Float16)(acc[mt][nt][i] + bn);
                *(uint4*)(C + (size_t)n * M + mbase) = fr.u;
            } else {
                float* C = (float*)Cout;
#pragma unroll
                for (int i = 0; i < 8; ++i) {
                    size_t idx = (size_t)(mbase + i) * N + n;
                    C[idx] = resid[idx] + acc[mt][nt][i] + bn;
                }
            }
        }
    }
}

// ---------------------------------------------------------------------------
// RoPE (in place on f16, rotate-half pairs (d, d+32)); cos/sin are (S,64) fp32.
// ---------------------------------------------------------------------------
__global__ __launch_bounds__(256) void rope_kernel(_Float16* __restrict__ buf,
                                                   const float* __restrict__ cosb,
                                                   const float* __restrict__ sinb,
                                                   int nheads) {
    const int t = blockIdx.x * blockDim.x + threadIdx.x;
    const int d = t & 31;
    const int hh = (t >> 5) % nheads;
    const int s = t / (32 * nheads);
    const size_t base = (size_t)s * nheads * HDIM + hh * HDIM + d;
    const float c = cosb[s * HDIM + d];
    const float sn = sinb[s * HDIM + d];
    const float a = (float)buf[base];
    const float b = (float)buf[base + 32];
    buf[base]      = (_Float16)(a * c - b * sn);
    buf[base + 32] = (_Float16)(b * c + a * sn);
}

// ---------------------------------------------------------------------------
// Flash attention, causal, GQA 4:1. One block = 128 queries of one head
// (8 waves x 16 rows). K/V tiles for each 32-key block are staged once into
// LDS with async copies and shared by all 8 waves. Computes S^T = K.Q^T so
// softmaxed P lands directly in the A-fragment layout for P.V.
// ---------------------------------------------------------------------------
__global__ __launch_bounds__(256) void attn_kernel(const _Float16* __restrict__ Q,
                                                   const _Float16* __restrict__ Kc,
                                                   const _Float16* __restrict__ Vt,
                                                   _Float16* __restrict__ O) {
    __shared__ _Float16 Ksm[32][72];   // 32 keys x 64 hd
    __shared__ _Float16 Vsm[64][40];   // 64 hd  x 32 keys (from Vt)
    const int t = threadIdx.x;
    const int lane = t & 31;
    const int wav = t >> 5;
    const int h = blockIdx.x >> 4;     // 16 q-blocks of 128 per head
    const int qblk0 = (blockIdx.x & 15) * 128;
    const int q0 = qblk0 + wav * 16;
    const int hkv = h >> 2;
    const int r = lane & 15;
    const int hi = (lane < 16) ? 0 : 8;
    const int m_g = q0 + r;            // query row owned by this lane

    // Staging coordinates (one 16B async copy each for K and V per thread).
    const int krow = t >> 3, kkc = (t & 7) * 8;   // K: 32 rows x 64 halves
    const int vrow = t >> 2, vkc = (t & 3) * 8;   // V: 64 rows x 32 halves

    // Q as B-fragments (lane = query column of Q^T), loaded once.
    F16Frag qf[2];
    const _Float16* qp = Q + (size_t)m_g * DMODEL + h * HDIM;
    qf[0].u[0] = *(const uint4*)(qp + hi);
    qf[0].u[1] = *(const uint4*)(qp + hi + 16);
    qf[1].u[0] = *(const uint4*)(qp + 32 + hi);
    qf[1].u[1] = *(const uint4*)(qp + 32 + hi + 16);

    v8f o[4];
#pragma unroll
    for (int nt = 0; nt < 4; ++nt) o[nt] = vzero();
    float runm = -1e30f, runl = 0.0f;

    for (int kb = 0; kb < qblk0 + 128; kb += 32) {
        // Stage K (32x64) and Vt (64x32) tiles for this key block.
        async_b128(&Ksm[krow][kkc],
                   Kc + (size_t)(kb + krow) * (NKV * HDIM) + hkv * HDIM + kkc);
        async_b128(&Vsm[vrow][vkc],
                   Vt + (size_t)(hkv * HDIM + vrow) * S_LEN + kb + vkc);
        wait_async0();
        __syncthreads();

        if (kb < q0 + 16) {            // wave-uniform: EXEC stays all-ones
            F16Frag ka[2][2];
#pragma unroll
            for (int kt = 0; kt < 2; ++kt) {
                ka[kt][0].u[0] = *(const uint4*)&Ksm[kt * 16 + r][hi];
                ka[kt][0].u[1] = *(const uint4*)&Ksm[kt * 16 + r][hi + 16];
                ka[kt][1].u[0] = *(const uint4*)&Ksm[kt * 16 + r][32 + hi];
                ka[kt][1].u[1] = *(const uint4*)&Ksm[kt * 16 + r][32 + hi + 16];
            }
            v8f s0 = vzero(), s1 = vzero();
            s0 = wmma16(ka[0][0], qf[0], s0);
            s0 = wmma16(ka[0][1], qf[1], s0);
            s1 = wmma16(ka[1][0], qf[0], s1);
            s1 = wmma16(ka[1][1], qf[1], s1);

            // scale + causal mask + row max (S^T: lane = query, VGPR = key)
            float mx = -1e30f;
#pragma unroll
            for (int i = 0; i < 8; ++i) {
                const int k0g = kb + i + hi;
                const int k1g = kb + 16 + i + hi;
                float v0 = (k0g <= m_g) ? s0[i] * 0.125f : NEG_V;
                float v1 = (k1g <= m_g) ? s1[i] * 0.125f : NEG_V;
                s0[i] = v0; s1[i] = v1;
                mx = fmaxf(mx, fmaxf(v0, v1));
            }
            mx = fmaxf(mx, __shfl_xor(mx, 16));   // lane pair = one query row
            const float newm = fmaxf(runm, mx);
            const float corr = __expf(runm - newm);

            float lsum = 0.0f;
            F16Frag pa;                            // P already in A-frag layout
#pragma unroll
            for (int i = 0; i < 8; ++i) {
                float e0 = __expf(s0[i] - newm);
                float e1 = __expf(s1[i] - newm);
                lsum += e0 + e1;
                pa.h[i] = (_Float16)e0;
                pa.h[8 + i] = (_Float16)e1;
            }
            runl = runl * corr + lsum;
            runm = newm;

            // Rescale O: factor for O-row (VGPR j) lives in lane j / j+8.
#pragma unroll
            for (int j = 0; j < 8; ++j) {
                float f = __shfl(corr, (lane < 16) ? j : (j + 8));
                o[0][j] *= f; o[1][j] *= f; o[2][j] *= f; o[3][j] *= f;
            }

            // P @ V from LDS (Vt rows are V columns -> contiguous chunks).
#pragma unroll
            for (int nt = 0; nt < 4; ++nt) {
                F16Frag vb;
                vb.u[0] = *(const uint4*)&Vsm[nt * 16 + r][hi];
                vb.u[1] = *(const uint4*)&Vsm[nt * 16 + r][hi + 16];
                o[nt] = wmma16(pa, vb, o[nt]);
            }
        }
        __syncthreads();   // protect LDS tiles before next staging round
    }

    const float l = runl + __shfl_xor(runl, 16);
    const float inv = 1.0f / l;
#pragma unroll
    for (int j = 0; j < 8; ++j) {
        float g = __shfl(inv, (lane < 16) ? j : (j + 8));
        const size_t rowoff = (size_t)(q0 + j + hi) * DMODEL + h * HDIM;
#pragma unroll
        for (int nt = 0; nt < 4; ++nt)
            O[rowoff + nt * 16 + r] = (_Float16)(o[nt][j] * g);
    }
}

// ---------------------------------------------------------------------------
// Fused gate/up GEMM + SiLU*mul: H[M,FFN] = silu(A@W1^T+b1) * (A@W3^T+b3), f16.
// Tile 128x64, 8 waves (4x2), wave tile 32x32 per weight matrix.
// Double-buffered; async A staging; W1/W3 through VGPR convert.
// ---------------------------------------------------------------------------
__global__ __launch_bounds__(256) void gateup_kernel(const _Float16* __restrict__ A,
                                                     const float* __restrict__ W1,
                                                     const float* __restrict__ B1,
                                                     const float* __restrict__ W3,
                                                     const float* __restrict__ B3,
                                                     _Float16* __restrict__ H,
                                                     int K) {
    __shared__ _Float16 Asm[2][128][40];
    __shared__ _Float16 Gsm[2][64][40];
    __shared__ _Float16 Usm[2][64][40];
    const int t = threadIdx.x;
    const int lane = t & 31;
    const int wav = t >> 5;
    const int wm = wav >> 1;    // 0..3
    const int wn = wav & 1;     // 0..1
    const int m0 = blockIdx.y * 128;
    const int n0 = blockIdx.x * 64;
    const int r  = lane & 15;
    const int hi = (lane < 16) ? 0 : 8;

    const int arow = t >> 2, akc = (t & 3) * 8;   // + i*64, 2 chunks
    const int brow = t >> 3, bkc = (t & 7) * 4;   // + i*32, 2 chunks per matrix

    v8f ag[2][2], au[2][2];
#pragma unroll
    for (int mt = 0; mt < 2; ++mt)
#pragma unroll
        for (int nt = 0; nt < 2; ++nt) { ag[mt][nt] = vzero(); au[mt][nt] = vzero(); }

    const int nk = K >> 5;

    // Prologue: tile 0 -> buffer 0.
#pragma unroll
    for (int i = 0; i < 2; ++i)
        async_b128(&Asm[0][arow + i*64][akc],
                   A + (size_t)(m0 + arow + i*64) * K + akc);
#pragma unroll
    for (int i = 0; i < 2; ++i) {
        float4 f1 = *(const float4*)(W1 + (size_t)(n0 + brow + i*32) * K + bkc);
        float4 f3 = *(const float4*)(W3 + (size_t)(n0 + brow + i*32) * K + bkc);
        h4 g, u;
        g[0] = (_Float16)f1.x; g[1] = (_Float16)f1.y; g[2] = (_Float16)f1.z; g[3] = (_Float16)f1.w;
        u[0] = (_Float16)f3.x; u[1] = (_Float16)f3.y; u[2] = (_Float16)f3.z; u[3] = (_Float16)f3.w;
        *(h4*)&Gsm[0][brow + i*32][bkc] = g;
        *(h4*)&Usm[0][brow + i*32][bkc] = u;
    }
    wait_async0();
    __syncthreads();

    for (int kk = 0; kk < nk; ++kk) {
        const int cur = kk & 1;
        const int nxt = cur ^ 1;
        const bool more = (kk + 1) < nk;
        const int k1 = (kk + 1) << 5;

        F16Frag af[2], gf[2], uf[2];
#pragma unroll
        for (int mt = 0; mt < 2; ++mt) {
            int row = wm * 32 + mt * 16 + r;
            af[mt].u[0] = *(const uint4*)&Asm[cur][row][hi];
            af[mt].u[1] = *(const uint4*)&Asm[cur][row][hi + 16];
        }
#pragma unroll
        for (int nt = 0; nt < 2; ++nt) {
            int row = wn * 32 + nt * 16 + r;
            gf[nt].u[0] = *(const uint4*)&Gsm[cur][row][hi];
            gf[nt].u[1] = *(const uint4*)&Gsm[cur][row][hi + 16];
            uf[nt].u[0] = *(const uint4*)&Usm[cur][row][hi];
            uf[nt].u[1] = *(const uint4*)&Usm[cur][row][hi + 16];
        }

        float4 f1r[2], f3r[2];
        if (more) {
#pragma unroll
            for (int i = 0; i < 2; ++i)
                async_b128(&Asm[nxt][arow + i*64][akc],
                           A + (size_t)(m0 + arow + i*64) * K + k1 + akc);
#pragma unroll
            for (int i = 0; i < 2; ++i) {
                f1r[i] = *(const float4*)(W1 + (size_t)(n0 + brow + i*32) * K + k1 + bkc);
                f3r[i] = *(const float4*)(W3 + (size_t)(n0 + brow + i*32) * K + k1 + bkc);
            }
        }

#pragma unroll
        for (int mt = 0; mt < 2; ++mt)
#pragma unroll
            for (int nt = 0; nt < 2; ++nt) {
                ag[mt][nt] = wmma16(af[mt], gf[nt], ag[mt][nt]);
                au[mt][nt] = wmma16(af[mt], uf[nt], au[mt][nt]);
            }

        if (more) {
#pragma unroll
            for (int i = 0; i < 2; ++i) {
                h4 g, u;
                g[0] = (_Float16)f1r[i].x; g[1] = (_Float16)f1r[i].y;
                g[2] = (_Float16)f1r[i].z; g[3] = (_Float16)f1r[i].w;
                u[0] = (_Float16)f3r[i].x; u[1] = (_Float16)f3r[i].y;
                u[2] = (_Float16)f3r[i].z; u[3] = (_Float16)f3r[i].w;
                *(h4*)&Gsm[nxt][brow + i*32][bkc] = g;
                *(h4*)&Usm[nxt][brow + i*32][bkc] = u;
            }
        }
        wait_async0();
        __syncthreads();
    }

#pragma unroll
    for (int mt = 0; mt < 2; ++mt) {
#pragma unroll
        for (int nt = 0; nt < 2; ++nt) {
            const int n = n0 + wn * 32 + nt * 16 + r;
            const float b1 = B1[n];
            const float b3 = B3[n];
            const int mbase = m0 + wm * 32 + mt * 16 + hi;
#pragma unroll
            for (int i = 0; i < 8; ++i) {
                float g = ag[mt][nt][i] + b1;
                float u = au[mt][nt][i] + b3;
                float sv = g / (1.0f + __expf(-g));
                H[(size_t)(mbase + i) * FFN_DIM + n] = (_Float16)(sv * u);
            }
        }
    }
}

// ---------------------------------------------------------------------------
extern "C" void kernel_launch(void* const* d_in, const int* in_sizes, int n_in,
                              void* d_out, int out_size, void* d_ws, size_t ws_size,
                              hipStream_t stream) {
    const float* hidden = (const float*)d_in[0];
    const float* cosb   = (const float*)d_in[1];
    const float* sinb   = (const float*)d_in[2];
    // d_in[3] attention_mask: causal, reconstructed in-kernel
    const float* wq  = (const float*)d_in[4];
    const float* bq  = (const float*)d_in[5];
    const float* wk  = (const float*)d_in[6];
    const float* bk  = (const float*)d_in[7];
    const float* wv  = (const float*)d_in[8];
    const float* bv  = (const float*)d_in[9];
    const float* wo  = (const float*)d_in[10];
    const float* bo  = (const float*)d_in[11];
    const float* ln1 = (const float*)d_in[12];
    const float* ln2 = (const float*)d_in[13];
    // d_in[14/15] gate: reference discards router output -> skipped
    const float* w1  = (const float*)d_in[16];
    const float* b1  = (const float*)d_in[17];
    const float* w2  = (const float*)d_in[18];
    const float* b2  = (const float*)d_in[19];
    const float* w3  = (const float*)d_in[20];
    const float* b3  = (const float*)d_in[21];
    float* out = (float*)d_out;

    char* ws = (char*)d_ws;
    _Float16* xnorm = (_Float16*)(ws);                       //  8 MB  S*D f16
    _Float16* qbuf  = (_Float16*)(ws + 8388608);             //  8 MB
    _Float16* kbuf  = (_Float16*)(ws + 16777216);            //  2 MB
    _Float16* vt    = (_Float16*)(ws + 18874368);            //  2 MB  (512 x S)
    _Float16* attnb = (_Float16*)(ws + 20971520);            //  8 MB
    _Float16* x2    = (_Float16*)(ws + 29360128);            //  8 MB
    _Float16* hact  = (_Float16*)(ws + 37748736);            // 32 MB  S*FFN f16

    // 1. RMSNorm 1
    rmsnorm_kernel<<<S_LEN, 256, 0, stream>>>(hidden, ln1, xnorm);
    // 2. QKV projections (V stored transposed)
    gemm_kernel<0><<<dim3(16, 16), 256, 0, stream>>>(xnorm, wq, bq, qbuf, nullptr, 2048, 2048);
    gemm_kernel<0><<<dim3(4, 16),  256, 0, stream>>>(xnorm, wk, bk, kbuf, nullptr, 512, 2048);
    gemm_kernel<1><<<dim3(4, 16),  256, 0, stream>>>(xnorm, wv, bv, vt,   nullptr, 512, 2048);
    // 3. RoPE on Q and K
    rope_kernel<<<(S_LEN * NHEADS * 32) / 256, 256, 0, stream>>>(qbuf, cosb, sinb, NHEADS);
    rope_kernel<<<(S_LEN * NKV * 32) / 256,    256, 0, stream>>>(kbuf, cosb, sinb, NKV);
    // 4. Flash attention (causal, GQA), LDS-shared K/V tiles
    attn_kernel<<<NHEADS * (S_LEN / 128), 256, 0, stream>>>(qbuf, kbuf, vt, attnb);
    // 5. O-proj + residual -> d_out (fp32)
    gemm_kernel<2><<<dim3(16, 16), 256, 0, stream>>>(attnb, wo, bo, out, hidden, 2048, 2048);
    // 6. RMSNorm 2
    rmsnorm_kernel<<<S_LEN, 256, 0, stream>>>(out, ln2, x2);
    // 7. Fused gate/up + SiLU*mul -> f16
    gateup_kernel<<<dim3(FFN_DIM / 64, 16), 256, 0, stream>>>(x2, w1, b1, w3, b3, hact, 2048);
    // 8. Down-proj + residual (d_out += mlp)
    gemm_kernel<2><<<dim3(16, 16), 256, 0, stream>>>(hact, w2, b2, out, out, 2048, 8192);
}
